// GraphCrossModule_57097295233744
// MI455X (gfx1250) — matrix operand order repaired and checked by reading, not compiled, not run on hardware
//
#include <hip/hip_runtime.h>
#include <hip/hip_bf16.h>

#define DF   128        // feature dim
#define ROWS 80         // GEMM rows per block (N = 50000 = 625 * 80)

typedef __attribute__((ext_vector_type(16))) __bf16 v16bf;
typedef __attribute__((ext_vector_type(8)))  __bf16 v8bf;
typedef __attribute__((ext_vector_type(8)))  float  v8f;
typedef __attribute__((ext_vector_type(4)))  unsigned int u32x4;
typedef __attribute__((ext_vector_type(4)))  int    i32x4;
typedef __attribute__((ext_vector_type(8)))  int    i32x8;

#if __has_builtin(__builtin_amdgcn_tensor_load_to_lds)
#define HAVE_TDM 1
#else
#define HAVE_TDM 0
#endif

__device__ __forceinline__ unsigned short f2bf(float f) {
    unsigned int x = __float_as_uint(f);
    unsigned int r = x + 0x7FFFu + ((x >> 16) & 1u);   // round-nearest-even
    return (unsigned short)(r >> 16);
}
__device__ __forceinline__ float sigf(float x) { return 1.0f / (1.0f + __expf(-x)); }
__device__ __forceinline__ unsigned int fmap(float f) {
    unsigned int u = __float_as_uint(f);
    return (u & 0x80000000u) ? ~u : (u | 0x80000000u); // order-preserving map
}

struct SelState { unsigned int prefix, mask, kneed, cnt; };

// ---------------- elementwise converts ----------------
__global__ void k_cvt_bf16(const float* __restrict__ s, unsigned short* __restrict__ d, long long n) {
    long long i = (long long)blockIdx.x * blockDim.x + threadIdx.x;
    if (i < n) d[i] = f2bf(s[i]);
}
// weights: convert + transpose -> Wt[n][k] bf16 (so LDS image is a straight 2D copy)
__global__ void k_cvt_w_t(const float* __restrict__ s, unsigned short* __restrict__ d) {
    int idx = blockIdx.x * blockDim.x + threadIdx.x;
    if (idx < DF * DF) { int k = idx >> 7, n = idx & 127; d[n * DF + k] = f2bf(s[idx]); }
}

// ---------------- degree / normalization ----------------
__global__ void k_deg_init(float* deg, int n) {
    int i = blockIdx.x * blockDim.x + threadIdx.x;
    if (i < n) deg[i] = 1.0f;                      // self-loop
}
__global__ void k_deg_count(const int* __restrict__ dst, float* deg, long long E) {
    long long e = (long long)blockIdx.x * blockDim.x + threadIdx.x;
    if (e < E) atomicAdd(&deg[dst[e]], 1.0f);
}
__global__ void k_inv_sqrt(float* deg, int n) {
    int i = blockIdx.x * blockDim.x + threadIdx.x;
    if (i < n) deg[i] = rsqrtf(fmaxf(deg[i], 1.0f));
}

// ---------------- WMMA GEMM: Y = X(bf16, n x 128) @ W(128x128) ----------------
// Wt is pre-transposed bf16: Wt[n*128+k]. Block: 256 thr = 8 wave32, 80 rows.
// Wave w owns cols [16w,16w+16), 5 row-tiles -> 20 v_wmma per wave.
// MODE 0: Yf = X@W.  MODE 1: Yb = bf16(prelu(X@W + bias)).
template <int MODE>
__global__ __launch_bounds__(256) void k_gemm(const unsigned short* __restrict__ Xb,
                                              const unsigned short* __restrict__ Wt,
                                              const float* __restrict__ bias,
                                              const float* __restrict__ prelu_a,
                                              float* __restrict__ Yf,
                                              unsigned short* __restrict__ Yb,
                                              int nrows) {
    __shared__ unsigned short lds_w[128 * 136];   // Wt with 8-elem pad per 128-elem row
    __shared__ unsigned short lds_x[ROWS * 136];
    const int t    = threadIdx.x;
    const int wave = t >> 5;
    const int lane = t & 31;
    const int m0   = blockIdx.x * ROWS;

#if HAVE_TDM
    // ---- TDM: DMA the 128x128 bf16 weight tile into LDS with hardware padding.
    // D# per ISA §8.3/8.4: count=1, type=2(image), data_size=2B, tile=tensor=128x128,
    // dim0_stride=128, pad_enable, pad_interval=64 DWORDs(=5), pad_amount=4 DWORDs(=3)
    // -> LDS row pitch 136 elements, exactly the layout the fragments below read.
    if (wave == 0) {
        unsigned long long ga = (unsigned long long)(uintptr_t)Wt;
        u32x4 g0;
        g0[0] = 1u;                                            // count=1
        g0[1] = (unsigned int)(uintptr_t)&lds_w[0];            // lds_addr
        g0[2] = (unsigned int)(ga & 0xFFFFFFFFu);              // global_addr lo
        g0[3] = (unsigned int)((ga >> 32) & 0x01FFFFFFu) | (2u << 30); // addr hi | type=2
        i32x8 g1;
        g1[0] = (1 << 16) | (1 << 20) | (5 << 22) | (3 << 25); // 2B elems, pad 4dw/64dw
        g1[1] = (int)(128u << 16);                             // tensor_dim0 = 128
        g1[2] = (int)(128u << 16);                             // tensor_dim1 = 128
        g1[3] = (int)(128u << 16);                             // tile_dim0   = 128
        g1[4] = 128;                                           // tile_dim1   = 128
        g1[5] = 128;                                           // tensor_dim0_stride
        g1[6] = 0; g1[7] = 0;
        i32x4 z4 = {};
#if defined(__clang_major__) && __clang_major__ >= 23
        i32x8 z8 = {};
        __builtin_amdgcn_tensor_load_to_lds(g0, g1, z4, z4, z8, 0);
#else
        __builtin_amdgcn_tensor_load_to_lds(g0, g1, z4, z4, 0);
#endif
    }
#else
    for (int idx = t; idx < 128 * 128; idx += 256) {
        int n = idx >> 7, k = idx & 127;
        lds_w[n * 136 + k] = Wt[idx];
    }
#endif

    // stage X tile (overlaps with the TDM transfer): 80x128, 16B per load
    for (int idx = t * 8; idx < ROWS * DF; idx += 2048) {
        int row = idx >> 7, cb = idx & 127;
        if (m0 + row < nrows)
            *(uint4*)(&lds_x[row * 136 + cb]) =
                *(const uint4*)(Xb + (size_t)(m0 + row) * DF + cb);
    }
#if HAVE_TDM
    if (wave == 0) __builtin_amdgcn_s_wait_tensorcnt(0);
#endif
    __syncthreads();

    const int nb   = wave * 16;
    const int half = lane >> 4;     // ISA §7.12.2: lanes 16-31 hold K 8-15 and 24-31 of each 32-K step
    const int mr   = lane & 15;
    v8f c[5] = {};
    #pragma unroll
    for (int ks = 0; ks < 4; ++ks) {
        const int kb = ks * 32;
        union { v16bf v; v8bf h[2]; } B;
        // B 32x16: lane holds col nb+mr; K runs kb+8h..+7 and kb+16+8h..+7 in lds_w
        B.h[0] = *(const v8bf*)(&lds_w[(nb + mr) * 136 + kb + 8 * half]);
        B.h[1] = *(const v8bf*)(&lds_w[(nb + mr) * 136 + kb + 16 + 8 * half]);
        #pragma unroll
        for (int rt = 0; rt < 5; ++rt) {
            union { v16bf v; v8bf h[2]; } A;
            const int mrow = rt * 16 + mr;
            A.h[0] = *(const v8bf*)(&lds_x[mrow * 136 + kb + 8 * half]);
            A.h[1] = *(const v8bf*)(&lds_x[mrow * 136 + kb + 16 + 8 * half]);
            c[rt] = __builtin_amdgcn_wmma_f32_16x16x32_bf16(false, A.v, false, B.v,
                                                            (short)0, c[rt], false, false);
        }
    }

    const int n = nb + mr;
    float a = 0.0f, bia = 0.0f;
    if (MODE == 1) { a = prelu_a[0]; bia = bias[n]; }
    #pragma unroll
    for (int rt = 0; rt < 5; ++rt) {
        #pragma unroll
        for (int r = 0; r < 8; ++r) {             // C/D: VGPR r -> row r+8*half
            const int m = m0 + rt * 16 + r + 8 * half;
            if (m >= nrows) continue;
            float v = c[rt][r];
            if (MODE == 1) {
                v += bia;
                v = (v >= 0.0f) ? v : a * v;
                Yb[(size_t)m * DF + n] = f2bf(v);
            } else {
                Yf[(size_t)m * DF + n] = v;
            }
        }
    }
}

// ---------------- GCN aggregation ----------------
__global__ void k_agg_init(float* __restrict__ out, const float* __restrict__ xw,
                           const float* __restrict__ inv, const float* __restrict__ bias,
                           const float* __restrict__ extra, long long total) {
    long long g = (long long)blockIdx.x * blockDim.x + threadIdx.x;
    if (g >= total) return;
    long long i = g >> 7; int j = (int)(g & 127);
    float c = inv[i];
    float v = bias[j] + c * c * xw[g];
    if (extra) v += extra[g];
    out[g] = v;
}
// one wave32 per edge; xw rows (25.6MB) are L2-resident on 192MB L2
__global__ void k_agg_edges(float* __restrict__ out, const float* __restrict__ xw,
                            const int* __restrict__ src, const int* __restrict__ dst,
                            const float* __restrict__ inv, long long E) {
    long long g = (long long)blockIdx.x * blockDim.x + threadIdx.x;
    long long e = g >> 5;
    int lane = (int)(g & 31);
    if (e >= E) return;
    int s = src[e], d = dst[e];
    float coef = inv[s] * inv[d];
    float4 v = ((const float4*)(xw + (size_t)s * DF))[lane];
    float* o = out + (size_t)d * DF + lane * 4;
    atomicAdd(o + 0, coef * v.x); atomicAdd(o + 1, coef * v.y);
    atomicAdd(o + 2, coef * v.z); atomicAdd(o + 3, coef * v.w);
}

// ---------------- bilinear score ----------------
__global__ void k_score(const float* __restrict__ t, const float* __restrict__ embed,
                        const float* __restrict__ bbil, float* __restrict__ scores, int n) {
    int row  = blockIdx.x * 8 + (threadIdx.x >> 5);
    int lane = threadIdx.x & 31;
    if (row >= n) return;
    float4 tv = ((const float4*)(t     + (size_t)row * DF))[lane];
    float4 ev = ((const float4*)(embed + (size_t)row * DF))[lane];
    float s = tv.x * sigf(ev.x) + tv.y * sigf(ev.y) + tv.z * sigf(ev.z) + tv.w * sigf(ev.w);
    #pragma unroll
    for (int off = 16; off > 0; off >>= 1) s += __shfl_xor(s, off, 32);
    if (lane == 0) scores[row] = sigf(s + bbil[0]);
}

// ---------------- radix top-k select ----------------
__global__ void k_sel_init(unsigned int* hist, SelState* st, unsigned int k) {
    int g = blockIdx.x * blockDim.x + threadIdx.x;
    if (g < 1024) hist[g] = 0u;
    if (g == 0) { st->prefix = 0u; st->mask = 0u; st->kneed = k; st->cnt = 0u; }
}
__global__ void k_hist(const float* __restrict__ scores, const SelState* __restrict__ st,
                       unsigned int* __restrict__ hist, int shift, int n) {
    int i = blockIdx.x * blockDim.x + threadIdx.x;
    if (i >= n) return;
    unsigned int u = fmap(scores[i]);
    if ((u & st->mask) == st->prefix) atomicAdd(&hist[(u >> shift) & 255u], 1u);
}
__global__ void k_resolve(const unsigned int* __restrict__ hist, SelState* st, int shift) {
    if (threadIdx.x != 0) return;
    unsigned int cum = 0, kneed = st->kneed;
    for (int b = 255; b >= 0; --b) {
        unsigned int h = hist[b];
        if (cum + h >= kneed) {
            st->kneed  = kneed - cum;
            st->prefix |= ((unsigned int)b) << shift;
            st->mask   |= 0xFFu << shift;
            return;
        }
        cum += h;
    }
}
__global__ void k_mask(const float* __restrict__ scores, SelState* st,
                       int* __restrict__ mask, int n) {
    int i = blockIdx.x * blockDim.x + threadIdx.x;
    if (i >= n) return;
    unsigned int u = fmap(scores[i]), thr = st->prefix;
    int m = 0;
    if (u > thr) m = 1;
    else if (u == thr) m = (atomicAdd(&st->cnt, 1u) < st->kneed) ? 1 : 0;
    mask[i] = m;
}

// ---------------- fine = mask ? gcn2 * score : 0 (bf16 for next GEMM) ----------------
__global__ void k_fine(const float* __restrict__ g2, const float* __restrict__ scores,
                       const int* __restrict__ mask, unsigned short* __restrict__ fineb,
                       long long total) {
    long long g = (long long)blockIdx.x * blockDim.x + threadIdx.x;
    if (g >= total) return;
    long long i = g >> 7;
    float v = mask[i] ? g2[g] * scores[i] : 0.0f;
    fineb[g] = f2bf(v);
}

// =====================================================================================
extern "C" void kernel_launch(void* const* d_in, const int* in_sizes, int n_in,
                              void* d_out, int out_size, void* d_ws, size_t ws_size,
                              hipStream_t stream) {
    const float* feat    = (const float*)d_in[0];
    const int*   ei      = (const int*)  d_in[2];
    const float* Wd      = (const float*)d_in[3];
    const float* bd      = (const float*)d_in[4];
    const float* prelu_a = (const float*)d_in[5];
    const float* Wbil    = (const float*)d_in[6];
    const float* bbil    = (const float*)d_in[7];
    const float* Wg1     = (const float*)d_in[8];
    const float* bg1     = (const float*)d_in[9];
    const float* Wg2     = (const float*)d_in[10];
    const float* bg2     = (const float*)d_in[11];
    const float* Wg3     = (const float*)d_in[12];
    const float* bg3     = (const float*)d_in[13];

    const int       N  = in_sizes[0] / DF;
    const long long E  = in_sizes[2] / 2;
    const int*      es = ei;
    const int*      ed = ei + E;
    const unsigned int K = (unsigned int)(N / 2);     // POOL_RATIO = 0.5
    const long long ND = (long long)N * DF;

    // -------- workspace carve-out (256B aligned) --------
    char* w = (char*)d_ws; size_t o = 0;
    auto alloc = [&](size_t bytes) { void* p = w + o; o = (o + bytes + 255) & ~(size_t)255; return p; };
    unsigned short* featb  = (unsigned short*)alloc(ND * 2);
    unsigned short* hposb  = (unsigned short*)alloc(ND * 2);
    unsigned short* embedb = (unsigned short*)alloc(ND * 2);
    unsigned short* fineb  = (unsigned short*)alloc(ND * 2);
    unsigned short* Wdb    = (unsigned short*)alloc(DF * DF * 2);
    unsigned short* Wg1b   = (unsigned short*)alloc(DF * DF * 2);
    unsigned short* Wg2b   = (unsigned short*)alloc(DF * DF * 2);
    unsigned short* Wg3b   = (unsigned short*)alloc(DF * DF * 2);
    unsigned short* Wbilb  = (unsigned short*)alloc(DF * DF * 2);
    float* bufA   = (float*)alloc(ND * 4);
    float* bufB   = (float*)alloc(ND * 4);
    float* embed  = (float*)alloc(ND * 4);
    float* inv    = (float*)alloc((size_t)N * 4);
    float* scores = (float*)alloc((size_t)N * 4);
    int*   maskb  = (int*)  alloc((size_t)N * 4);
    unsigned int* hist = (unsigned int*)alloc(1024 * 4);
    SelState* st  = (SelState*)alloc(sizeof(SelState));
    float* outp   = (float*)d_out;

    const int TB = 256;
    dim3 b(TB);
    const int gND  = (int)((ND + TB - 1) / TB);
    const int gN   = (N + TB - 1) / TB;
    const int gE32 = (int)((E * 32 + TB - 1) / TB);
    const int gE   = (int)((E + TB - 1) / TB);
    const int gW   = (DF * DF + TB - 1) / TB;
    const int gM   = (N + ROWS - 1) / ROWS;           // 625 blocks for N=50000

    // 1) converts (weights transposed for column-major LDS image)
    k_cvt_bf16<<<gND, b, 0, stream>>>(feat, featb, ND);
    k_cvt_w_t <<<gW,  b, 0, stream>>>(Wd,   Wdb);
    k_cvt_w_t <<<gW,  b, 0, stream>>>(Wg1,  Wg1b);
    k_cvt_w_t <<<gW,  b, 0, stream>>>(Wg2,  Wg2b);
    k_cvt_w_t <<<gW,  b, 0, stream>>>(Wg3,  Wg3b);
    k_cvt_w_t <<<gW,  b, 0, stream>>>(Wbil, Wbilb);

    // 2) symmetric normalization coefficients
    k_deg_init <<<gN, b, 0, stream>>>(inv, N);
    k_deg_count<<<gE, b, 0, stream>>>(ed, inv, E);
    k_inv_sqrt <<<gN, b, 0, stream>>>(inv, N);

    // 3) h_pos = prelu(feat @ Wd + bd)   (bf16 out; h_neg path is dead code)
    k_gemm<1><<<gM, b, 0, stream>>>(featb, Wdb, bd, prelu_a, nullptr, hposb, N);

    // 4) embed = GCN1(h_pos)
    k_gemm<0>  <<<gM,  b, 0, stream>>>(hposb, Wg1b, nullptr, nullptr, bufA, nullptr, N);
    k_agg_init <<<gND, b, 0, stream>>>(embed, bufA, inv, bg1, nullptr, ND);
    k_agg_edges<<<gE32,b, 0, stream>>>(embed, bufA, es, ed, inv, E);
    k_cvt_bf16 <<<gND, b, 0, stream>>>(embed, embedb, ND);

    // 5) scores = sigmoid( (h_pos @ Wbil) . sigmoid(embed) + bbil )
    k_gemm<0><<<gM, b, 0, stream>>>(hposb, Wbilb, nullptr, nullptr, bufB, nullptr, N);
    k_score  <<<(N + 7) / 8, b, 0, stream>>>(bufB, embed, bbil, scores, N);

    // 6) radix select of k-th largest score (argsort only decides membership)
    k_sel_init<<<4, b, 0, stream>>>(hist, st, K);
    for (int p = 0; p < 4; ++p) {
        int shift = 24 - 8 * p;
        k_hist   <<<gN, b, 0, stream>>>(scores, st, hist + p * 256, shift, N);
        k_resolve<<<1, 32, 0, stream>>>(hist + p * 256, st, shift);
    }

    // 7) gcn2 = GCN2(embed)
    k_gemm<0>  <<<gM,  b, 0, stream>>>(embedb, Wg2b, nullptr, nullptr, bufA, nullptr, N);
    k_agg_init <<<gND, b, 0, stream>>>(bufB, bufA, inv, bg2, nullptr, ND);
    k_agg_edges<<<gE32,b, 0, stream>>>(bufB, bufA, es, ed, inv, E);

    // 8) fine = mask ? gcn2 * score : 0
    k_mask<<<gN,  b, 0, stream>>>(scores, st, maskb, N);
    k_fine<<<gND, b, 0, stream>>>(bufB, scores, maskb, fineb, ND);

    // 9) out = GCN3(fine) + embed   (skip-add fused into agg_init)
    k_gemm<0>  <<<gM,  b, 0, stream>>>(fineb, Wg3b, nullptr, nullptr, bufA, nullptr, N);
    k_agg_init <<<gND, b, 0, stream>>>(outp, bufA, inv, bg3, embed, ND);
    k_agg_edges<<<gE32,b, 0, stream>>>(outp, bufA, es, ed, inv, E);
}
